// StarAttention_24756191494475
// MI455X (gfx1250) — matrix operand order repaired and checked by reading, not compiled, not run on hardware
//
#include <hip/hip_runtime.h>
#include <hip/hip_bf16.h>

// ---------------------------------------------------------------------------
// StarAttention (single query token, alpha-entmax) for MI455X / gfx1250.
//
// Algebraic restructuring (q is a single token):
//   scores[b,h,l] = (1/8) * ( (q_h^T Wk_h) . y[b,:,l] + q_h . bk_h )
//   u[b,h,c]      = sum_l entmax(scores)[b,h,l] * y[b,c,l]
//   att[b,hd]     = Wv[hd,:] . u[b,h,:] + bv[hd]          (since sum_l p = 1)
//   out           = att @ Wo^T + bo
// => two memory-bound streaming passes over y (2 x 256 MB ~ 22 us at 23.3 TB/s),
//    compute made free with v_wmma_f32_16x16x32_bf16 (M=16 padded, fp32 accum).
// Pass 1 stages y tiles into LDS with double-buffered
// global_load_async_to_lds_b128 (ASYNCcnt-pipelined DMA overlapping WMMA).
// ---------------------------------------------------------------------------

#define NB     16
#define NHID_  512
#define LL     8192
#define NHEAD_ 8
#define HD     64

typedef __attribute__((ext_vector_type(16))) __bf16 v16bf;
typedef __attribute__((ext_vector_type(8)))  __bf16 v8bf;
typedef __attribute__((ext_vector_type(8)))  float  v8f;

__device__ inline v16bf make_afrag(const __bf16* __restrict__ ap) {
    // A-matrix 16x32 bf16 layout: lane holds row M=lane&15, K-halves split by lane/16.
    // elements 0..7  <- K = kg*8 + 0..7     (contiguous 16B)
    // elements 8..15 <- K = kg*8 + 16..23   (contiguous 16B at +16 elems)
    v8bf lo = *(const v8bf*)ap;
    v8bf hi = *(const v8bf*)(ap + 16);
    v16bf a;
#pragma unroll
    for (int i = 0; i < 8; ++i) { a[i] = lo[i]; a[8 + i] = hi[i]; }
    return a;
}

__device__ inline v16bf cvt16_to_bf(float4 f0, float4 f1, float4 f2, float4 f3) {
    v16bf r;
    r[0]  = (__bf16)f0.x; r[1]  = (__bf16)f0.y; r[2]  = (__bf16)f0.z; r[3]  = (__bf16)f0.w;
    r[4]  = (__bf16)f1.x; r[5]  = (__bf16)f1.y; r[6]  = (__bf16)f1.z; r[7]  = (__bf16)f1.w;
    r[8]  = (__bf16)f2.x; r[9]  = (__bf16)f2.y; r[10] = (__bf16)f2.z; r[11] = (__bf16)f2.w;
    r[12] = (__bf16)f3.x; r[13] = (__bf16)f3.y; r[14] = (__bf16)f3.z; r[15] = (__bf16)f3.w;
    return r;
}

// ---------------------------------------------------------------------------
// K1: q[b,o] = x[b,:] . Wq[o,:] + bq[o]
__global__ __launch_bounds__(256)
void k_q(const float* __restrict__ x, const float* __restrict__ Wq,
         const float* __restrict__ bq, float* __restrict__ q) {
    int t = blockIdx.x * 256 + threadIdx.x;   // NB*NHID_ = 8192
    int b = t >> 9, o = t & 511;
    const float* xr = x + (size_t)b * NHID_;
    const float* wr = Wq + (size_t)o * NHID_;
    float s = bq[o];
#pragma unroll 4
    for (int c = 0; c < NHID_; ++c) s += xr[c] * wr[c];
    q[t] = s;
}

// K2: Aqk[b][m][c] (bf16, M padded to 16): rows 0..7 = 0.125 * q_h^T Wk_h, rows 8..15 = 0
__global__ __launch_bounds__(256)
void k_qk(const float* __restrict__ q, const float* __restrict__ Wk,
          __bf16* __restrict__ Aqk) {
    int t = blockIdx.x * 256 + threadIdx.x;   // NB*NHEAD_*NHID_ = 65536
    int b = t >> 12;
    int h = (t >> 9) & 7;
    int c = t & 511;
    const float* qr = q + (size_t)b * NHID_ + h * HD;
    float s = 0.f;
#pragma unroll 4
    for (int d = 0; d < HD; ++d) s += qr[d] * Wk[(size_t)(h * HD + d) * NHID_ + c];
    Aqk[((size_t)b * 16 + h)     * NHID_ + c] = (__bf16)(0.125f * s);
    Aqk[((size_t)b * 16 + 8 + h) * NHID_ + c] = (__bf16)0.0f;
}

// K2b: qoff[b,h] = 0.125 * q_h . bk_h   (bk bias folded into scores)
__global__ __launch_bounds__(128)
void k_qoff(const float* __restrict__ q, const float* __restrict__ bk,
            float* __restrict__ qoff) {
    int t = threadIdx.x;
    if (t >= NB * NHEAD_) return;
    int b = t >> 3, h = t & 7;
    const float* qr = q + (size_t)b * NHID_ + h * HD;
    float s = 0.f;
#pragma unroll
    for (int d = 0; d < HD; ++d) s += qr[d] * bk[h * HD + d];
    qoff[t] = 0.125f * s;
}

// ---------------------------------------------------------------------------
// K3: streaming pass 1 — scores[b,h,l] via WMMA.
// Per block: batch b, 256-wide l-tile, 128 threads = 4 waves, each wave 64 cols.
// y tile (32 ch x 256 l, fp32) staged in natural [c][l] order into LDS with
// double-buffered global_load_async_to_lds_b128 (16 asyncs/wave/step, in-order
// completion lets "s_wait_asynccnt 16" release the previous buffer while the
// next tile's DMA runs under the WMMA work). The K-transpose happens in the
// LDS->VGPR gather (ds_load_b32).
#define CSTRIDE 260   // floats per channel row in LDS (mult of 4 -> 16B rows)

__global__ __launch_bounds__(128)
void k_scores(const float* __restrict__ y, const __bf16* __restrict__ Aqk,
              const float* __restrict__ qoff, const unsigned char* __restrict__ mask,
              float* __restrict__ scores) {
    __shared__ __align__(16) float tile[2][32][CSTRIDE];

    const int b    = blockIdx.x;
    const int lt   = blockIdx.y * 256;
    const int tid  = threadIdx.x;
    const int lane = tid & 31;
    const int wave = tid >> 5;
    const int m    = lane & 15;
    const int kg   = lane >> 4;

    v8f acc[4] = {};  // 4 n-subtiles of 16 columns

    const size_t ybase = (size_t)b * NHID_ * LL + lt;

    auto stage = [&](int buf, int c0) {
        // copy y[b, c0:c0+32, lt:lt+256] (fp32) -> tile[buf][c][l], b128 DMA
#pragma unroll
        for (int j = 0; j < 16; ++j) {
            int g  = tid + 128 * j;        // float4 index within 32x256 tile
            int c  = g >> 6;               // 64 float4 per channel row
            int l4 = (g & 63) << 2;
            const float* gp = y + ybase + (size_t)(c0 + c) * LL + l4;
            unsigned int lo = (unsigned int)(size_t)&tile[buf][c][l4];
            asm volatile("global_load_async_to_lds_b128 %0, %1, off"
                         :: "v"(lo), "v"(gp) : "memory");
        }
    };

    stage(0, 0);
    int buf = 0;
    for (int c0 = 0; c0 < NHID_; c0 += 32) {
        if (c0 + 32 < NHID_) {
            stage(buf ^ 1, c0 + 32);                     // prefetch next tile
            asm volatile("s_wait_asynccnt 16" ::: "memory");  // prev 16 done
        } else {
            asm volatile("s_wait_asynccnt 0" ::: "memory");
        }
        __syncthreads();

        v16bf afrag = make_afrag(Aqk + (size_t)(b * 16 + m) * NHID_ + c0 + kg * 8);

#pragma unroll
        for (int nt = 0; nt < 4; ++nt) {
            int l = wave * 64 + nt * 16 + m;   // this lane's B-matrix column
            v16bf bfrag;
#pragma unroll
            for (int e = 0; e < 16; ++e)       // gather K=channel (transpose)
                bfrag[e] = (__bf16)tile[buf][kg * 8 + e + (e >= 8 ? 8 : 0)][l];
            acc[nt] = __builtin_amdgcn_wmma_f32_16x16x32_bf16(
                          false, afrag, false, bfrag, (short)0, acc[nt], false, false);
        }
        __syncthreads();
        buf ^= 1;
    }

    // C layout: VGPR r, lanes 0..15 -> M=r (heads 0..7 valid), col = lane
    if (lane < 16) {
#pragma unroll
        for (int nt = 0; nt < 4; ++nt) {
            int l = lt + wave * 64 + nt * 16 + lane;
            bool msk = mask[(size_t)b * LL + l] != 0;
#pragma unroll
            for (int h = 0; h < 8; ++h) {
                float s = msk ? -100.0f : (acc[nt][h] + qoff[b * NHEAD_ + h]);
                scores[((size_t)b * NHEAD_ + h) * LL + l] = s;
            }
        }
    }
}

// ---------------------------------------------------------------------------
// K4: alpha-entmax via 50-step bisection; one block per (b,h) row of 8192.
// Writes normalized weights as bf16 into padded A matrix Aal[b][16][L]
// (row h = weights, row 8+h = zeros for the WMMA pad).
__global__ __launch_bounds__(256)
void k_entmax(const float* __restrict__ scores, const float* __restrict__ alpha,
              __bf16* __restrict__ Aal) {
    const int b = blockIdx.x >> 3;
    const int h = blockIdx.x & 7;
    const int tid = threadIdx.x;
    __shared__ float red[8];

    float a = alpha[h];
    a = fminf(fmaxf(a, 1.0f), 2.0f);
    const float am1 = a - 1.0f;
    const float inv = 1.0f / am1;

    const float* row = scores + ((size_t)b * NHEAD_ + h) * LL;
    float xa[32];
    float mx = -1e30f;
#pragma unroll
    for (int j = 0; j < 32; ++j) {
        float v = row[j * 256 + tid] * am1;
        xa[j] = v;
        mx = fmaxf(mx, v);
    }
#pragma unroll
    for (int off = 16; off; off >>= 1) mx = fmaxf(mx, __shfl_xor(mx, off, 32));
    if ((tid & 31) == 0) red[tid >> 5] = mx;
    __syncthreads();
    float gmx = red[0];
#pragma unroll
    for (int i = 1; i < 8; ++i) gmx = fmaxf(gmx, red[i]);
    __syncthreads();

    float tau_lo = gmx - 1.0f;
    float dm = (gmx - __powf(1.0f / (float)LL, am1)) - tau_lo;
    float tau_m = tau_lo;

    for (int it = 0; it < 50; ++it) {
        dm *= 0.5f;
        tau_m = tau_lo + dm;
        float s = 0.0f;
#pragma unroll
        for (int j = 0; j < 32; ++j) {
            float t = xa[j] - tau_m;
            s += (t > 0.0f) ? __powf(t, inv) : 0.0f;
        }
#pragma unroll
        for (int off = 16; off; off >>= 1) s += __shfl_xor(s, off, 32);
        if ((tid & 31) == 0) red[tid >> 5] = s;
        __syncthreads();
        float tot = 0.0f;
#pragma unroll
        for (int i = 0; i < 8; ++i) tot += red[i];
        __syncthreads();
        if (tot - 1.0f >= 0.0f) tau_lo = tau_m;   // block-uniform, no divergence
    }

    float p[32];
    float s = 0.0f;
#pragma unroll
    for (int j = 0; j < 32; ++j) {
        float t = xa[j] - tau_m;
        p[j] = (t > 0.0f) ? __powf(t, inv) : 0.0f;
        s += p[j];
    }
#pragma unroll
    for (int off = 16; off; off >>= 1) s += __shfl_xor(s, off, 32);
    if ((tid & 31) == 0) red[tid >> 5] = s;
    __syncthreads();
    float tot = 0.0f;
#pragma unroll
    for (int i = 0; i < 8; ++i) tot += red[i];
    const float r = 1.0f / tot;

    __bf16* out  = Aal + ((size_t)b * 16 + h) * LL;
    __bf16* outz = Aal + ((size_t)b * 16 + 8 + h) * LL;
#pragma unroll
    for (int j = 0; j < 32; ++j) {
        out[j * 256 + tid]  = (__bf16)(p[j] * r);
        outz[j * 256 + tid] = (__bf16)0.0f;
    }
}

// ---------------------------------------------------------------------------
// K5: streaming pass 2 — u[b,h,c] = sum_l p[b,h,l] * y[b,c,l] via WMMA.
// K (=l) is contiguous in y, so B fragments load straight from global (two 32B
// runs per lane). Grid splits K into 8 chunks; partials merged with
// atomicAdd. Unroll-by-2 keeps more loads in flight between WMMAs.
__global__ __launch_bounds__(128)
void k_uacc(const float* __restrict__ y, const __bf16* __restrict__ Aal,
            float* __restrict__ u) {
    const int b  = blockIdx.x;
    const int ct = blockIdx.y * 128;          // channel tile base (N)
    const int kc = blockIdx.z * 1024;         // K chunk over L
    const int tid = threadIdx.x;
    const int lane = tid & 31;
    const int wave = tid >> 5;
    const int m  = lane & 15;
    const int kg = lane >> 4;

    v8f acc[2] = {};
    const int cbase = ct + wave * 32;

#pragma unroll 2
    for (int k0 = kc; k0 < kc + 1024; k0 += 32) {
        v16bf afrag = make_afrag(Aal + (size_t)(b * 16 + m) * LL + k0 + kg * 8);
#pragma unroll
        for (int nt = 0; nt < 2; ++nt) {
            int c = cbase + nt * 16 + m;      // B-matrix column = channel
            const float* yp = y + ((size_t)b * NHID_ + c) * LL + k0 + kg * 8;
            __builtin_prefetch(yp + 256, 0, 1);
            const float4* yp4 = (const float4*)yp;
            v16bf bfrag = cvt16_to_bf(yp4[0], yp4[1], yp4[4], yp4[5]);
            acc[nt] = __builtin_amdgcn_wmma_f32_16x16x32_bf16(
                          false, afrag, false, bfrag, (short)0, acc[nt], false, false);
        }
    }

    if (lane < 16) {
#pragma unroll
        for (int nt = 0; nt < 2; ++nt) {
            int c = cbase + nt * 16 + lane;
#pragma unroll
            for (int h = 0; h < 8; ++h)
                atomicAdd(&u[((size_t)b * NHEAD_ + h) * NHID_ + c], acc[nt][h]);
        }
    }
}

// ---------------------------------------------------------------------------
// K6a: att[b,hd] = Wv[hd,:] . u[b,h,:] + bv[hd]
__global__ __launch_bounds__(256)
void k_att(const float* __restrict__ u, const float* __restrict__ Wv,
           const float* __restrict__ bv, float* __restrict__ att) {
    int t = blockIdx.x * 256 + threadIdx.x;   // NB*512
    int b = t >> 9, hd = t & 511, h = hd >> 6;
    const float* ur = u + ((size_t)b * NHEAD_ + h) * NHID_;
    const float* wr = Wv + (size_t)hd * NHID_;
    float s = bv[hd];
#pragma unroll 4
    for (int c = 0; c < NHID_; ++c) s += wr[c] * ur[c];
    att[t] = s;
}

// K6b: out[b,o] = att[b,:] . Wo[o,:] + bo[o]
__global__ __launch_bounds__(256)
void k_out(const float* __restrict__ att, const float* __restrict__ Wo,
           const float* __restrict__ bo, float* __restrict__ out) {
    int t = blockIdx.x * 256 + threadIdx.x;   // NB*512
    int b = t >> 9, o = t & 511;
    const float* ar = att + (size_t)b * NHID_;
    const float* wr = Wo + (size_t)o * NHID_;
    float s = bo[o];
#pragma unroll 4
    for (int hd = 0; hd < NHID_; ++hd) s += ar[hd] * wr[hd];
    out[t] = s;
}

__global__ __launch_bounds__(256)
void k_zero(float* __restrict__ p, int n) {
    int t = blockIdx.x * 256 + threadIdx.x;
    if (t < n) p[t] = 0.0f;
}

// ---------------------------------------------------------------------------
extern "C" void kernel_launch(void* const* d_in, const int* in_sizes, int n_in,
                              void* d_out, int out_size, void* d_ws, size_t ws_size,
                              hipStream_t stream) {
    const float* x     = (const float*)d_in[0];
    const float* y     = (const float*)d_in[1];
    const float* Wq    = (const float*)d_in[2];
    const float* bq    = (const float*)d_in[3];
    const float* Wk    = (const float*)d_in[4];
    const float* bk    = (const float*)d_in[5];
    const float* Wv    = (const float*)d_in[6];
    const float* bv    = (const float*)d_in[7];
    const float* Wo    = (const float*)d_in[8];
    const float* bo    = (const float*)d_in[9];
    const float* alpha = (const float*)d_in[10];
    const unsigned char* mask = (const unsigned char*)d_in[11];

    char* ws = (char*)d_ws;
    auto alloc = [&](size_t bytes) {
        char* p = ws;
        ws += (bytes + 255) & ~(size_t)255;
        return p;
    };
    float*  q      = (float*)alloc((size_t)NB * NHID_ * 4);
    float*  qoff   = (float*)alloc((size_t)NB * NHEAD_ * 4);
    __bf16* Aqk    = (__bf16*)alloc((size_t)NB * 16 * NHID_ * 2);
    float*  scores = (float*)alloc((size_t)NB * NHEAD_ * LL * 4);
    __bf16* Aal    = (__bf16*)alloc((size_t)NB * 16 * LL * 2);
    float*  u      = (float*)alloc((size_t)NB * NHEAD_ * NHID_ * 4);
    float*  att    = (float*)alloc((size_t)NB * NHID_ * 4);

    k_q   <<<32, 256, 0, stream>>>(x, Wq, bq, q);
    k_qk  <<<256, 256, 0, stream>>>(q, Wk, Aqk);
    k_qoff<<<1, 128, 0, stream>>>(q, bk, qoff);

    dim3 g3(NB, LL / 256);
    k_scores<<<g3, 128, 0, stream>>>(y, Aqk, qoff, mask, scores);

    k_entmax<<<NB * NHEAD_, 256, 0, stream>>>(scores, alpha, Aal);

    int un = NB * NHEAD_ * NHID_;
    k_zero<<<(un + 255) / 256, 256, 0, stream>>>(u, un);

    dim3 g5(NB, NHID_ / 128, LL / 1024);
    k_uacc<<<g5, 128, 0, stream>>>(y, Aal, u);

    k_att<<<32, 256, 0, stream>>>(u, Wv, bv, att);
    k_out<<<32, 256, 0, stream>>>(att, Wo, bo, (float*)d_out);
}
